// Attention_79688823210495
// MI455X (gfx1250) — matrix operand order repaired and checked by reading, not compiled, not run on hardware
//
#include <hip/hip_runtime.h>
#include <hip/hip_bf16.h>

// ---------------- CDNA5 WMMA types ----------------
typedef __attribute__((ext_vector_type(16))) __bf16   v16bf;
typedef __attribute__((ext_vector_type(8)))  float    v8f;
typedef __attribute__((ext_vector_type(8)))  unsigned v8u;

__device__ __forceinline__ v8f vzero8() {
    v8f z = {0.f,0.f,0.f,0.f,0.f,0.f,0.f,0.f};
    return z;
}

__device__ __forceinline__ v8f wmma_bf16(v16bf a, v16bf b, v8f c) {
    // D = A(16x32 bf16) * B(32x16 bf16) + C(16x16 f32)
    return __builtin_amdgcn_wmma_f32_16x16x32_bf16(
        /*neg_a=*/false, a, /*neg_b=*/false, b,
        /*c_mod=*/(short)0, c, /*reuse_a=*/false, /*reuse_b=*/false);
}

// f32 -> bf16 round-to-nearest-even
__device__ __forceinline__ unsigned short f2bf(float f) {
    unsigned x = __builtin_bit_cast(unsigned, f);
    unsigned r = x + 0x7FFFu + ((x >> 16) & 1u);
    return (unsigned short)(r >> 16);
}

// A-fragment loader (ISA 7.12.2, 16-bit A 16x32, wave32):
//   lane L: row M = base + (L&15); half = L>>4
//   VGPR p<4 : K = 2p     + 8*half   (bytes 4p      + 16*half)
//   VGPR p>=4: K = 16+2(p-4)+8*half  (bytes 32+4(p-4)+16*half)
__device__ __forceinline__ v16bf load_frag_a(const unsigned short* src,
                                             int row, int ld, int k0) {
    int lane  = threadIdx.x & 31;
    int half8 = (lane >> 4) << 3;
    const char* base = (const char*)(src + (size_t)row * ld + k0);
    union { v8u u; v16bf b; } f;
#pragma unroll
    for (int p = 0; p < 8; ++p) {
        int kb = ((p < 4) ? (p * 2) : (16 + (p - 4) * 2)) + half8;
        f.u[p] = *(const unsigned*)(base + kb * 2);
    }
    return f.b;
}

// B-fragment loader (ISA 7.12.4 B-matrix striping, 16-bit, wave32):
//   lane L: column N = base + (L&15); half = L>>4
//   VGPR v<4: K = 2v + 8*half; v>=4: K = 16+2(v-4)+8*half (same as A)
__device__ __forceinline__ v16bf load_frag_b(const unsigned short* src,
                                             int row, int ld, int k0) {
    int lane = threadIdx.x & 31;
    int half16 = (lane >> 4) << 3;
    const unsigned* p = (const unsigned*)(src + (size_t)row * ld + k0 + half16);
    union { v8u u; v16bf b; } f;
#pragma unroll
    for (int q = 0; q < 8; ++q) f.u[q] = p[(q < 4) ? q : (q + 4)];
    return f.b;
}

// butterfly reductions within 16-lane halves (rows live per half)
__device__ __forceinline__ float rowmax16(float v) {
#pragma unroll
    for (int m = 8; m >= 1; m >>= 1) v = fmaxf(v, __shfl_xor(v, m, 16));
    return v;
}
__device__ __forceinline__ float rowsum16(float v) {
#pragma unroll
    for (int m = 8; m >= 1; m >>= 1) v += __shfl_xor(v, m, 16);
    return v;
}

// ---------------- problem constants ----------------
#define BATCH   16
#define NTOK    1024
#define INPD    512
#define HEADS   8
#define DHEAD   32
#define INNER   256      // HEADS*DHEAD
#define QKV3    768      // 3*INNER
#define OUPD    512
#define MTOT    16384    // BATCH*NTOK

// ---------------- prep: f32 -> bf16 conversions ----------------
__global__ void cvt_x_kernel(const float* __restrict__ x,
                             unsigned short* __restrict__ xb, int n) {
    for (int i = blockIdx.x * blockDim.x + threadIdx.x; i < n;
         i += gridDim.x * blockDim.x)
        xb[i] = f2bf(x[i]);
}

__global__ void tr_wqkv_kernel(const float* __restrict__ w,   // [512][768]
                               unsigned short* __restrict__ wt) { // [768][512]
    int i = blockIdx.x * blockDim.x + threadIdx.x;
    if (i < INPD * QKV3) {
        int kk = i / QKV3, c = i % QKV3;
        wt[(size_t)c * INPD + kk] = f2bf(w[i]);
    }
}

__global__ void tr_wout_kernel(const float* __restrict__ w,   // [256][512]
                               unsigned short* __restrict__ wt) { // [512][256]
    int i = blockIdx.x * blockDim.x + threadIdx.x;
    if (i < INNER * OUPD) {
        int kk = i / OUPD, c = i % OUPD;
        wt[(size_t)c * INNER + kk] = f2bf(w[i]);
    }
}

// ---------------- QKV GEMM: [16384,512] x [512,768] ----------------
// block = 256 thr (8 waves, 4x2), tile 64(M) x 128(N); wave tile 16x64
// scatter epilogue -> q[bh][n][d], k[bh][n][d], vT[bh][d][n]  (bf16)
__global__ void gemm_qkv_kernel(const unsigned short* __restrict__ xb,
                                const unsigned short* __restrict__ wt,
                                unsigned short* __restrict__ qb,
                                unsigned short* __restrict__ kbuf,
                                unsigned short* __restrict__ vt) {
    int lane = threadIdx.x & 31, wave = threadIdx.x >> 5;
    int half = lane >> 4, l15 = lane & 15;
    int m0 = blockIdx.x * 64 + (wave >> 1) * 16;
    int n0 = blockIdx.y * 128 + (wave & 1) * 64;

    v8f acc[4];
#pragma unroll
    for (int j = 0; j < 4; ++j) acc[j] = vzero8();

    for (int k0 = 0; k0 < INPD; k0 += 32) {
        v16bf a = load_frag_a(xb, m0 + l15, INPD, k0);
#pragma unroll
        for (int j = 0; j < 4; ++j) {
            v16bf b = load_frag_b(wt, n0 + j * 16 + l15, INPD, k0);
            acc[j] = wmma_bf16(a, b, acc[j]);
        }
    }
#pragma unroll
    for (int j = 0; j < 4; ++j) {
        int c = n0 + j * 16 + l15;
        int seg = c >> 8, cc = c & 255, hh = cc >> 5, d = cc & 31;
#pragma unroll
        for (int r = 0; r < 8; ++r) {
            int m = m0 + r + half * 8;
            int bi = m >> 10, nn = m & 1023;
            unsigned short val = f2bf(acc[j][r]);
            size_t bh = (size_t)(bi * HEADS + hh);
            if (seg == 0)      qb[(bh * NTOK + nn) * DHEAD + d] = val;
            else if (seg == 1) kbuf[(bh * NTOK + nn) * DHEAD + d] = val;
            else               vt[(bh * DHEAD + d) * NTOK + nn] = val;
        }
    }
}

// ---------------- flash attention ----------------
// grid (128 bh, 8), block 256 (8 waves); wave owns 16 query rows.
__global__ void attn_kernel(const unsigned short* __restrict__ qb,
                            const unsigned short* __restrict__ kb,
                            const unsigned short* __restrict__ vt,
                            const float* __restrict__ bias_table,
                            unsigned short* __restrict__ ao) {
    __shared__ __align__(16) unsigned short plds[8][16 * 36];
    int lane = threadIdx.x & 31, wave = threadIdx.x >> 5;
    int half = lane >> 4, l15 = lane & 15;
    int bh = blockIdx.x, h = bh & 7;
    int qrow0 = (blockIdx.y * 8 + wave) * 16;

    const unsigned short* qbh = qb + (size_t)bh * NTOK * DHEAD;
    const unsigned short* kbh = kb + (size_t)bh * NTOK * DHEAD;
    const unsigned short* vth = vt + (size_t)bh * DHEAD * NTOK;

    v16bf qa = load_frag_a(qbh, qrow0 + l15, DHEAD, 0);

    v8f acc0 = vzero8(), acc1 = vzero8();
    float mrow[8], lrow[8];
    int irow[8], icol[8];
#pragma unroll
    for (int r = 0; r < 8; ++r) {
        mrow[r] = -3.0e38f; lrow[r] = 0.f;
        int i = qrow0 + r + half * 8;
        irow[r] = i >> 5; icol[r] = i & 31;
    }
    const float scale = 0.17677669529663687f;  // 32^-0.5

    for (int j0 = 0; j0 < NTOK; j0 += 32) {
        v16bf kf0 = load_frag_b(kbh, j0 + l15, DHEAD, 0);
        v16bf kf1 = load_frag_b(kbh, j0 + 16 + l15, DHEAD, 0);
        v8f s0 = wmma_bf16(qa, kf0, vzero8());
        v8f s1 = wmma_bf16(qa, kf1, vzero8());

        int ja = j0 + l15, jb = ja + 16;
        int jra = ja >> 5, jca = ja & 31, jrb = jb >> 5, jcb = jb & 31;
        float p0[8], p1[8], alpha[8];
#pragma unroll
        for (int r = 0; r < 8; ++r) {
            int ia = ((irow[r] - jra + 31) * 63 + (icol[r] - jca + 31));
            int ib = ((irow[r] - jrb + 31) * 63 + (icol[r] - jcb + 31));
            float a  = s0[r] * scale + bias_table[(ia << 3) + h];
            float b2 = s1[r] * scale + bias_table[(ib << 3) + h];
            float mx = rowmax16(fmaxf(a, b2));
            float mn = fmaxf(mrow[r], mx);
            alpha[r] = __expf(mrow[r] - mn);
            p0[r] = __expf(a - mn);
            p1[r] = __expf(b2 - mn);
            lrow[r] = lrow[r] * alpha[r] + rowsum16(p0[r] + p1[r]);
            mrow[r] = mn;
        }
#pragma unroll
        for (int r = 0; r < 8; ++r) { acc0[r] *= alpha[r]; acc1[r] *= alpha[r]; }

        // C-layout P -> LDS -> A-layout fragment (same-wave LDS ops in-order)
        unsigned short* pl = plds[wave];
#pragma unroll
        for (int r = 0; r < 8; ++r) {
            int row = r + half * 8;
            pl[row * 36 + l15]      = f2bf(p0[r]);
            pl[row * 36 + 16 + l15] = f2bf(p1[r]);
        }
        v16bf pa  = load_frag_a(pl, l15, 36, 0);
        v16bf vb0 = load_frag_b(vth, l15, NTOK, j0);        // N = d 0..15
        v16bf vb1 = load_frag_b(vth, 16 + l15, NTOK, j0);   // N = d 16..31
        acc0 = wmma_bf16(pa, vb0, acc0);
        acc1 = wmma_bf16(pa, vb1, acc1);
    }

    int bi = bh >> 3;
#pragma unroll
    for (int r = 0; r < 8; ++r) {
        int i = qrow0 + r + half * 8;
        float inv = 1.0f / lrow[r];
        size_t base = ((size_t)(bi * NTOK + i)) * INNER + h * DHEAD;
        ao[base + l15]      = f2bf(acc0[r] * inv);
        ao[base + 16 + l15] = f2bf(acc1[r] * inv);
    }
}

// ---------------- output GEMM: [16384,256] x [256,512] + bias ----------------
__global__ void gemm_out_kernel(const unsigned short* __restrict__ ao,
                                const unsigned short* __restrict__ wt, // [512][256]
                                const float* __restrict__ b_out,
                                float* __restrict__ out) {
    int lane = threadIdx.x & 31, wave = threadIdx.x >> 5;
    int half = lane >> 4, l15 = lane & 15;
    int m0 = blockIdx.x * 64 + (wave >> 1) * 16;
    int n0 = blockIdx.y * 128 + (wave & 1) * 64;

    v8f acc[4];
#pragma unroll
    for (int j = 0; j < 4; ++j) acc[j] = vzero8();

    for (int k0 = 0; k0 < INNER; k0 += 32) {
        v16bf a = load_frag_a(ao, m0 + l15, INNER, k0);
#pragma unroll
        for (int j = 0; j < 4; ++j) {
            v16bf b = load_frag_b(wt, n0 + j * 16 + l15, INNER, k0);
            acc[j] = wmma_bf16(a, b, acc[j]);
        }
    }
#pragma unroll
    for (int j = 0; j < 4; ++j) {
        int c = n0 + j * 16 + l15;
        float bias = b_out[c];
#pragma unroll
        for (int r = 0; r < 8; ++r) {
            int m = m0 + r + half * 8;
            out[(size_t)m * OUPD + c] = acc[j][r] + bias;
        }
    }
}

// ---------------- launcher ----------------
extern "C" void kernel_launch(void* const* d_in, const int* in_sizes, int n_in,
                              void* d_out, int out_size, void* d_ws, size_t ws_size,
                              hipStream_t stream) {
    const float* x          = (const float*)d_in[0];  // [16,1024,512]
    const float* w_qkv      = (const float*)d_in[1];  // [512,768]
    const float* bias_table = (const float*)d_in[2];  // [3969,8]
    const float* w_out      = (const float*)d_in[3];  // [256,512]
    const float* b_out      = (const float*)d_in[4];  // [512]
    float* out = (float*)d_out;

    char* ws = (char*)d_ws;
    size_t off = 0;
    unsigned short* xb    = (unsigned short*)(ws + off); off += (size_t)MTOT * INPD * 2;   // 16 MB
    unsigned short* wqkvT = (unsigned short*)(ws + off); off += (size_t)QKV3 * INPD * 2;   // 768 KB
    unsigned short* woutT = (unsigned short*)(ws + off); off += (size_t)OUPD * INNER * 2;  // 256 KB
    unsigned short* qb    = (unsigned short*)(ws + off); off += (size_t)BATCH*HEADS*NTOK*DHEAD*2;
    unsigned short* kb    = (unsigned short*)(ws + off); off += (size_t)BATCH*HEADS*NTOK*DHEAD*2;
    unsigned short* vt    = (unsigned short*)(ws + off); off += (size_t)BATCH*HEADS*DHEAD*NTOK*2;
    unsigned short* ao    = (unsigned short*)(ws + off); off += (size_t)MTOT * INNER * 2;
    // total ~49 MB

    cvt_x_kernel<<<4096, 256, 0, stream>>>(x, xb, MTOT * INPD);
    tr_wqkv_kernel<<<(INPD * QKV3 + 255) / 256, 256, 0, stream>>>(w_qkv, wqkvT);
    tr_wout_kernel<<<(INNER * OUPD + 255) / 256, 256, 0, stream>>>(w_out, woutT);

    gemm_qkv_kernel<<<dim3(MTOT / 64, QKV3 / 128), 256, 0, stream>>>(xb, wqkvT, qb, kb, vt);

    attn_kernel<<<dim3(BATCH * HEADS, NTOK / 128), 256, 0, stream>>>(qb, kb, vt, bias_table, ao);

    gemm_out_kernel<<<dim3(MTOT / 64, OUPD / 128), 256, 0, stream>>>(ao, woutT, b_out, out);
}
